// EdgeBlock_5076651344271
// MI455X (gfx1250) — matrix operand order repaired and checked by reading, not compiled, not run on hardware
//
#include <hip/hip_runtime.h>

typedef __attribute__((ext_vector_type(16))) _Float16 v16h;
typedef __attribute__((ext_vector_type(8)))  _Float16 v8h;
typedef __attribute__((ext_vector_type(4)))  _Float16 v4h;
typedef __attribute__((ext_vector_type(8)))  float    v8f;

#define XP 264   // LDS row pitch in halves: 256 + 8 pad (16B) -> de-conflicts banks

// ---------------------------------------------------------------------------
// Pack a KxN row-major f32 weight matrix into f16 WMMA B-fragment order.
// B layout (16-bit, wave32): lane L -> col N = nt*16 + (L&15);
//   halves e=0..15 -> K = kt*32 + (e>>3)*16 + (L>>4)*8 + (e&7).
// dst[((nt*KT + kt)*32 + L)*16 + e]  (each fragment = 1KB contiguous).
// ---------------------------------------------------------------------------
__global__ void pack_w(const float* __restrict__ src, _Float16* __restrict__ dst,
                       int Kdim, int Ndim) {
    int p = blockIdx.x * 256 + threadIdx.x;
    if (p >= Kdim * Ndim) return;
    int e    = p & 15;
    int L    = (p >> 4) & 31;
    int frag = p >> 9;
    int KT   = Kdim >> 5;
    int kt   = frag % KT;
    int nt   = frag / KT;
    int N = nt * 16 + (L & 15);
    int K = kt * 32 + (e >> 3) * 16 + (L >> 4) * 8 + (e & 7);
    dst[p] = (_Float16)src[(long)K * Ndim + N];
}

// convert 4 f32 -> 4 f16 and store 8B to LDS
__device__ __forceinline__ void cvt4(_Float16* d, float4 v) {
    v4h o;
    o[0] = (_Float16)v.x; o[1] = (_Float16)v.y;
    o[2] = (_Float16)v.z; o[3] = (_Float16)v.w;
    *(v4h*)d = o;
}

__device__ __forceinline__ v8f wmma_f16(v16h a, v16h b, v8f c) {
    return __builtin_amdgcn_wmma_f32_16x16x32_f16(false, a, false, b,
                                                  (short)0, c, false, false);
}

// fast SiLU: v_exp + v_add + v_rcp + v_mul (avoid IEEE divide expansion)
__device__ __forceinline__ float silu(float x) {
    return x * __builtin_amdgcn_rcpf(1.0f + __expf(-x));
}

// ---------------------------------------------------------------------------
// Main fused kernel: one block = 256 edges, 8 waves, each wave owns 32 rows
// (two 16-row WMMA tiles -> each B fragment is loaded once, used twice).
// B fragments are streamed through a depth-2 register pipeline so the wmma
// chain never waits on a just-issued load. Single in-place activation buffer.
// ---------------------------------------------------------------------------
__global__ __launch_bounds__(256)
void edge_mlp(const float* __restrict__ pooled, const float* __restrict__ edges,
              const float* __restrict__ globs,  const float* __restrict__ cndts,
              const int*   __restrict__ bidx,
              const float* __restrict__ b1, const float* __restrict__ b2,
              const float* __restrict__ b3,
              const _Float16* __restrict__ W1p, const _Float16* __restrict__ W2p,
              const _Float16* __restrict__ W3p,
              float* __restrict__ out) {
    __shared__ _Float16 Xs[256 * XP];

    const int t    = threadIdx.x;
    const int lane = t & 31;
    const int w    = t >> 5;       // wave id 0..7
    const int hf   = lane >> 4;    // half-wave 0/1
    const int ln16 = lane & 15;
    const long rowBase = (long)blockIdx.x * 256;

    // ---- stage X tile: [256 rows x 256 cols] f32 -> f16 LDS ----
    #pragma unroll
    for (int g = 0; g < 2; g++) {
        int r = (t >> 1) + g * 128, h2 = t & 1;
        long gr = rowBase + r;
        _Float16* xr = Xs + r * XP;
        const float4* ps = (const float4*)(pooled + gr * 128 + h2 * 64);
        #pragma unroll
        for (int i = 0; i < 16; i++) cvt4(xr + h2 * 64 + i * 4, ps[i]);
        const float4* es = (const float4*)(edges + gr * 64 + h2 * 32);
        #pragma unroll
        for (int i = 0; i < 8; i++)  cvt4(xr + 128 + h2 * 32 + i * 4, es[i]);
        int b = bidx[gr];
        const float4* cs = (const float4*)(h2 ? (cndts + (long)b * 32)
                                              : (globs + (long)b * 32));
        #pragma unroll
        for (int i = 0; i < 8; i++)  cvt4(xr + 192 + h2 * 32 + i * 4, cs[i]);
    }
    __syncthreads();

    // A-fragment load (16-bit A 16x32 layout): two contiguous 16B pieces.
    auto load_afrag = [&](int mt, int kt) -> v16h {
        const _Float16* base = Xs + (w * 32 + mt * 16 + ln16) * XP + kt * 32 + hf * 8;
        v8h lo = *(const v8h*)base;
        v8h hi = *(const v8h*)(base + 16);
        v16h a;
        #pragma unroll
        for (int i = 0; i < 8; i++) { a[i] = lo[i]; a[i + 8] = hi[i]; }
        return a;
    };

    v16h A0[8], A1[8];

    // ---- layer body as a macro-ish lambda family ------------------------
    // GEMM over NT n-tiles with depth-2 pipelined B stream.
    // Epilogue handled inline per layer (LDS writeback vs global store).

    // ================= Layer 1: Xs(256) -> Xs(256) in place, SiLU ========
    #pragma unroll
    for (int k = 0; k < 8; k++) { A0[k] = load_afrag(0, k); A1[k] = load_afrag(1, k); }
    __syncthreads();                       // all reads done before overwrite
    {
        const _Float16* Wp = W1p;
        const int FMAX = 16 * 8;
        auto ldB = [&](int f) -> v16h {
            int ff = f < FMAX ? f : FMAX - 1;
            return *(const v16h*)(Wp + ((ff * 32 + lane) << 4));
        };
        v16h bA = ldB(0), bB = ldB(1);
        #pragma unroll 2
        for (int nt = 0; nt < 16; nt++) {
            float bias = b1[nt * 16 + ln16];
            v8f c0 = {}, c1 = {};
            #pragma unroll
            for (int k = 0; k < 8; k++) {
                v16h bn = ldB(nt * 8 + k + 2);
                c0 = wmma_f16(A0[k], bA, c0);
                c1 = wmma_f16(A1[k], bA, c1);
                bA = bB; bB = bn;
            }
            #pragma unroll
            for (int v = 0; v < 8; v++) {
                Xs[(w * 32 +      hf * 8 + v) * XP + nt * 16 + ln16] = (_Float16)silu(c0[v] + bias);
                Xs[(w * 32 + 16 + hf * 8 + v) * XP + nt * 16 + ln16] = (_Float16)silu(c1[v] + bias);
            }
        }
    }
    __syncthreads();

    // ================= Layer 2: Xs(256) -> Xs(256) in place, SiLU ========
    #pragma unroll
    for (int k = 0; k < 8; k++) { A0[k] = load_afrag(0, k); A1[k] = load_afrag(1, k); }
    __syncthreads();
    {
        const _Float16* Wp = W2p;
        const int FMAX = 16 * 8;
        auto ldB = [&](int f) -> v16h {
            int ff = f < FMAX ? f : FMAX - 1;
            return *(const v16h*)(Wp + ((ff * 32 + lane) << 4));
        };
        v16h bA = ldB(0), bB = ldB(1);
        #pragma unroll 2
        for (int nt = 0; nt < 16; nt++) {
            float bias = b2[nt * 16 + ln16];
            v8f c0 = {}, c1 = {};
            #pragma unroll
            for (int k = 0; k < 8; k++) {
                v16h bn = ldB(nt * 8 + k + 2);
                c0 = wmma_f16(A0[k], bA, c0);
                c1 = wmma_f16(A1[k], bA, c1);
                bA = bB; bB = bn;
            }
            #pragma unroll
            for (int v = 0; v < 8; v++) {
                Xs[(w * 32 +      hf * 8 + v) * XP + nt * 16 + ln16] = (_Float16)silu(c0[v] + bias);
                Xs[(w * 32 + 16 + hf * 8 + v) * XP + nt * 16 + ln16] = (_Float16)silu(c1[v] + bias);
            }
        }
    }
    __syncthreads();

    // ================= Layer 3: Xs(256) -> out[:,0:128] ==================
    #pragma unroll
    for (int k = 0; k < 8; k++) { A0[k] = load_afrag(0, k); A1[k] = load_afrag(1, k); }
    {
        const _Float16* Wp = W3p;
        const int FMAX = 8 * 8;
        auto ldB = [&](int f) -> v16h {
            int ff = f < FMAX ? f : FMAX - 1;
            return *(const v16h*)(Wp + ((ff * 32 + lane) << 4));
        };
        v16h bA = ldB(0), bB = ldB(1);
        #pragma unroll 2
        for (int nt = 0; nt < 8; nt++) {
            float bias = b3[nt * 16 + ln16];
            v8f c0 = {}, c1 = {};
            #pragma unroll
            for (int k = 0; k < 8; k++) {
                v16h bn = ldB(nt * 8 + k + 2);
                c0 = wmma_f16(A0[k], bA, c0);
                c1 = wmma_f16(A1[k], bA, c1);
                bA = bB; bB = bn;
            }
            #pragma unroll
            for (int v = 0; v < 8; v++) {
                long g0 = rowBase + w * 32 +      hf * 8 + v;
                long g1 = rowBase + w * 32 + 16 + hf * 8 + v;
                out[g0 * 192 + nt * 16 + ln16] = c0[v] + bias;
                out[g1 * 192 + nt * 16 + ln16] = c1[v] + bias;
            }
        }
    }

    // ---- residual 'cat': out[:,128:192] = edges ----
    #pragma unroll
    for (int g = 0; g < 2; g++) {
        int r = (t >> 1) + g * 128, h2 = t & 1;
        long gr = rowBase + r;
        const float4* es = (const float4*)(edges + gr * 64 + h2 * 32);
        float4*       od = (float4*)(out + gr * 192 + 128 + h2 * 32);
        #pragma unroll
        for (int i = 0; i < 8; i++) od[i] = es[i];
    }
}

// ---------------------------------------------------------------------------
extern "C" void kernel_launch(void* const* d_in, const int* in_sizes, int n_in,
                              void* d_out, int out_size, void* d_ws, size_t ws_size,
                              hipStream_t stream) {
    const float* pooled = (const float*)d_in[0];
    const float* edges  = (const float*)d_in[1];
    const float* globs  = (const float*)d_in[2];
    const float* cndts  = (const float*)d_in[3];
    const int*   bidx   = (const int*)  d_in[4];
    const float* W1 = (const float*)d_in[5];
    const float* b1 = (const float*)d_in[6];
    const float* W2 = (const float*)d_in[7];
    const float* b2 = (const float*)d_in[8];
    const float* W3 = (const float*)d_in[9];
    const float* b3 = (const float*)d_in[10];
    float* out = (float*)d_out;

    const int E = in_sizes[0] / 128;   // pooled_mssgs is [E,128]

    _Float16* wsh = (_Float16*)d_ws;
    _Float16* W1p = wsh;
    _Float16* W2p = wsh + 256 * 256;
    _Float16* W3p = wsh + 2 * 256 * 256;

    pack_w<<<(256 * 256) / 256, 256, 0, stream>>>(W1, W1p, 256, 256);
    pack_w<<<(256 * 256) / 256, 256, 0, stream>>>(W2, W2p, 256, 256);
    pack_w<<<(256 * 128) / 256, 256, 0, stream>>>(W3, W3p, 256, 128);

    edge_mlp<<<E / 256, 256, 0, stream>>>(pooled, edges, globs, cndts, bidx,
                                          b1, b2, b3, W1p, W2p, W3p, out);
}
